// BBoxAndClsDecoder_58892591563097
// MI455X (gfx1250) — matrix operand order repaired and checked by reading, not compiled, not run on hardware
//
#include <hip/hip_runtime.h>
#include <hip/hip_bf16.h>

// ---------------------------------------------------------------------------
// Problem constants
// ---------------------------------------------------------------------------
#define BB   8
#define TT   24
#define QQ   256
#define DD   512
#define CC   91
#define MTOT (BB * TT * QQ)        // 49152 rows
#define KDEPTH 512                 // all GEMMs have K = 512
#define KT_N  (KDEPTH / 32)        // 16 K-tiles
#define CHUNK 8192                 // rows per chunk (divides MTOT; multiple of 64)
#define NCHUNK (MTOT / CHUNK)
#define IMGW 1088.0f
#define IMGH 608.0f

// WMMA vector types (gfx1250, wave32)
typedef __bf16 v16bf __attribute__((ext_vector_type(16)));
typedef float  v8f   __attribute__((ext_vector_type(8)));

union BFrag {
    v16bf          v;
    unsigned short s[16];
    unsigned int   u[8];
    uint4          q[2];
};

// ---------------------------------------------------------------------------
// fp32 -> bf16 split helpers (round-to-nearest-even)
// ---------------------------------------------------------------------------
__device__ inline unsigned short f32_to_bf16_rne(float x) {
    unsigned int u = __float_as_uint(x);
    unsigned int r = u + 0x7FFFu + ((u >> 16) & 1u);
    return (unsigned short)(r >> 16);
}

__device__ inline void split_bf16(float x, unsigned short* hi, unsigned short* lo) {
    unsigned short h = f32_to_bf16_rne(x);
    float hf = __uint_as_float(((unsigned int)h) << 16);
    *hi = h;
    *lo = f32_to_bf16_rne(x - hf);
}

// ---------------------------------------------------------------------------
// Weight swizzle: fp32 W[K=512][Nsrc] -> bf16 hi/lo fragments laid out in the
// WMMA B-operand lane layout (n = lane%16) with contiguous k = 16*(lane/16) + s.
// dword index = (((kt*NT + nt)*2 + hilo)*32 + lane)*8 + j
// ---------------------------------------------------------------------------
__global__ void swizzle_weight_kernel(const float* __restrict__ W,
                                      unsigned int* __restrict__ dst,
                                      int Nsrc, int Npad) {
    int t = blockIdx.x * blockDim.x + threadIdx.x;
    int NT = Npad >> 4;
    int total = KT_N * NT * 32;
    if (t >= total) return;
    int lane = t & 31;
    int frag = t >> 5;              // frag = kt*NT + nt
    int nt = frag % NT;
    int n  = nt * 16 + (lane & 15);
    int kb = (frag / NT) * 32 + (lane >> 4) * 16;

    unsigned int hiw[8], low[8];
#pragma unroll
    for (int j = 0; j < 8; ++j) {
        float x0 = (n < Nsrc) ? W[(size_t)(kb + 2 * j) * Nsrc + n] : 0.0f;
        float x1 = (n < Nsrc) ? W[(size_t)(kb + 2 * j + 1) * Nsrc + n] : 0.0f;
        unsigned short h0, l0, h1, l1;
        split_bf16(x0, &h0, &l0);
        split_bf16(x1, &h1, &l1);
        hiw[j] = (unsigned int)h0 | ((unsigned int)h1 << 16);
        low[j] = (unsigned int)l0 | ((unsigned int)l1 << 16);
    }
    unsigned int* dhi = dst + ((size_t)(frag * 2 + 0) * 32 + lane) * 8;
    unsigned int* dlo = dst + ((size_t)(frag * 2 + 1) * 32 + lane) * 8;
#pragma unroll
    for (int j = 0; j < 8; ++j) { dhi[j] = hiw[j]; dlo[j] = low[j]; }
}

// ---------------------------------------------------------------------------
// Activation split: fp32 X[CHUNK x 512] -> bf16 hi/lo fragments pre-packed in
// the WMMA A-operand lane layout (m = lane%16) with contiguous k = 16*(lane/16) + s.
// dword index = (((mt*KT_N + kt)*2 + hilo)*32 + lane)*8 + j
// ---------------------------------------------------------------------------
__global__ void split_acts_kernel(const float* __restrict__ X,
                                  unsigned int* __restrict__ dst) {
    int t = blockIdx.x * blockDim.x + threadIdx.x;   // (CHUNK/16)*KT_N*32 threads
    int lane = t & 31;
    int frag = t >> 5;               // frag = mt*KT_N + kt
    int kt = frag & (KT_N - 1);
    int mt = frag >> 4;
    int row = mt * 16 + (lane & 15);
    int kb  = kt * 32 + (lane >> 4) * 16;
    const float* xr = X + (size_t)row * KDEPTH;

    float4 a0 = *(const float4*)(xr + kb);
    float4 a1 = *(const float4*)(xr + kb + 4);
    float4 a2 = *(const float4*)(xr + kb + 8);
    float4 a3 = *(const float4*)(xr + kb + 12);
    float av[16] = {a0.x, a0.y, a0.z, a0.w, a1.x, a1.y, a1.z, a1.w,
                    a2.x, a2.y, a2.z, a2.w, a3.x, a3.y, a3.z, a3.w};
    BFrag hi, lo;
#pragma unroll
    for (int j = 0; j < 16; ++j) split_bf16(av[j], &hi.s[j], &lo.s[j]);

    unsigned int* dhi = dst + ((size_t)(frag * 2 + 0) * 32 + lane) * 8;
    unsigned int* dlo = dst + ((size_t)(frag * 2 + 1) * 32 + lane) * 8;
    *(uint4*)(dhi)     = hi.q[0];
    *(uint4*)(dhi + 4) = hi.q[1];
    *(uint4*)(dlo)     = lo.q[0];
    *(uint4*)(dlo + 4) = lo.q[1];
}

// ---------------------------------------------------------------------------
// Fused GEMM + bias + activation:  Y = act(A[M x 512] * W[512 x NT*16] + b)
// A and W are pre-split/pre-swizzled bf16 hi/lo fragment buffers.
// bf16x3 emulation of fp32: C += Ahi*Bhi + Alo*Bhi + Ahi*Blo
// Template: NFRAG n-tiles per wave, WN waves along N (4 waves along M),
// MODE 0 = ReLU, 1 = sigmoid.  Fully unrolled, software-pipelined:
// A double-buffered across kt, B double-buffered across the fragment loop.
// ---------------------------------------------------------------------------
template <int NFRAG, int WN, int MODE>
__global__ __launch_bounds__(32 * 4 * WN) void gemm_bf16x3_t(
    const unsigned int* __restrict__ Asw,
    const unsigned int* __restrict__ Wsw,
    const float* __restrict__ bias,
    float* __restrict__ Y, int ldy, int NT, int Nact)
{
    const int tid  = threadIdx.x;
    const int lane = tid & 31;
    const int wv   = tid >> 5;
    const int m0  = __builtin_amdgcn_readfirstlane((int)blockIdx.x * 64 + (wv & 3) * 16);
    const int nt0 = __builtin_amdgcn_readfirstlane(((int)blockIdx.y * WN + (wv >> 2)) * NFRAG);
    const int mt  = m0 >> 4;

    // Fragment base pointers (uint4 units; one fragment slab = 128 uint4)
    const uint4* abase = (const uint4*)Asw + (size_t)mt * KT_N * 128 + lane * 2;
    const uint4* bbase = (const uint4*)Wsw + (size_t)nt0 * 128 + lane * 2;
    const int bstride = NT * 128;   // per-kt stride in uint4

    v8f c[NFRAG];
#pragma unroll
    for (int f = 0; f < NFRAG; ++f)
#pragma unroll
        for (int r = 0; r < 8; ++r) c[f][r] = 0.0f;

    BFrag ahi[2], alo[2], bhi[2], blo[2];

    auto LA = [&](int kt, int buf) {
        const uint4* ap = abase + kt * 128;
        ahi[buf].q[0] = ap[0];  ahi[buf].q[1] = ap[1];
        alo[buf].q[0] = ap[64]; alo[buf].q[1] = ap[65];
    };
    auto LB = [&](int kt, int f, int buf) {
        const uint4* bp = bbase + kt * bstride + f * 128;
        bhi[buf].q[0] = bp[0];  bhi[buf].q[1] = bp[1];
        blo[buf].q[0] = bp[64]; blo[buf].q[1] = bp[65];
    };

    LA(0, 0);
    LB(0, 0, 0);
#pragma unroll
    for (int kt = 0; kt < KT_N; ++kt) {
        const int cur = kt & 1;
        if (kt + 1 < KT_N) LA(kt + 1, cur ^ 1);
#pragma unroll
        for (int f = 0; f < NFRAG; ++f) {
            const int fb = (kt * NFRAG + f) & 1;
            // prefetch next B fragment while computing this one
            if (f + 1 < NFRAG)       LB(kt, f + 1, fb ^ 1);
            else if (kt + 1 < KT_N)  LB(kt + 1, 0, fb ^ 1);
            c[f] = __builtin_amdgcn_wmma_f32_16x16x32_bf16(
                       false, ahi[cur].v, false, bhi[fb].v, (short)0, c[f], false, false);
            c[f] = __builtin_amdgcn_wmma_f32_16x16x32_bf16(
                       false, alo[cur].v, false, bhi[fb].v, (short)0, c[f], false, false);
            c[f] = __builtin_amdgcn_wmma_f32_16x16x32_bf16(
                       false, ahi[cur].v, false, blo[fb].v, (short)0, c[f], false, false);
        }
    }

    // Epilogue: bias + activation + store (C/D layout: n=lane%16, m=r+(lane/16)*8)
#pragma unroll
    for (int f = 0; f < NFRAG; ++f) {
        const int n = (nt0 + f) * 16 + (lane & 15);
        if (n < Nact) {
            const float bv = bias[n];
#pragma unroll
            for (int r = 0; r < 8; ++r) {
                const int m = m0 + (lane >> 4) * 8 + r;
                float v = c[f][r] + bv;
                v = (MODE == 0) ? fmaxf(v, 0.0f) : 1.0f / (1.0f + __expf(-v));
                Y[(size_t)m * ldy + n] = v;
            }
        }
    }
}

// ---------------------------------------------------------------------------
// ori: cxcywh -> xyxy scaled to image size
// ---------------------------------------------------------------------------
__global__ void ori_kernel(const float* __restrict__ bbox,
                           float* __restrict__ ori) {
    int m = blockIdx.x * blockDim.x + threadIdx.x;
    if (m >= MTOT) return;
    float cx = bbox[m * 4 + 0], cy = bbox[m * 4 + 1];
    float w  = bbox[m * 4 + 2], h  = bbox[m * 4 + 3];
    ori[m * 4 + 0] = (cx - 0.5f * w) * IMGW;
    ori[m * 4 + 1] = (cy - 0.5f * h) * IMGH;
    ori[m * 4 + 2] = (cx + 0.5f * w) * IMGW;
    ori[m * 4 + 3] = (cy + 0.5f * h) * IMGH;
}

// ---------------------------------------------------------------------------
// scores = max over 91 classes per row
// ---------------------------------------------------------------------------
__global__ void scores_kernel(const float* __restrict__ cls,
                              float* __restrict__ scores) {
    int m = blockIdx.x * blockDim.x + threadIdx.x;
    if (m >= MTOT) return;
    const float* row = cls + (size_t)m * CC;
    float mx = row[0];
    for (int j = 1; j < CC; ++j) mx = fmaxf(mx, row[j]);
    scores[m] = mx;
}

// ---------------------------------------------------------------------------
// Sequential per-frame track-ID assignment: one block per batch, 256 threads
// (= Q queries = 8 wave32s). Prefix-sum of "new track" flags in query order.
// ---------------------------------------------------------------------------
__global__ void track_kernel(const float* __restrict__ scores,
                             int* __restrict__ out_idx) {
    const int b = blockIdx.x;
    const int q = threadIdx.x;
    const int lane = q & 31;
    const int wv = q >> 5;
    __shared__ int wsum[8];

    int obj = -1;
    int cnt = 0;   // replicated across threads; all add the same frame total
    for (int t = 0; t < TT; ++t) {
        float s = scores[((size_t)b * TT + t) * QQ + q];
        bool nw = (obj < 0) && (s >= 0.5f);
        unsigned int mask = (unsigned int)__ballot((int)nw);
        int pre = __popc(mask & ((1u << lane) - 1u));
        int tot = __popc(mask);
        if (lane == 0) wsum[wv] = tot;
        __syncthreads();
        int off = 0, total = 0;
        for (int i = 0; i < 8; ++i) { if (i < wv) off += wsum[i]; total += wsum[i]; }
        int incl = off + pre + (nw ? 1 : 0);
        if (nw) obj = cnt + incl - 1;
        cnt += total;
        out_idx[((size_t)b * TT + t) * QQ + q] = obj;
        __syncthreads();
    }
}

// ---------------------------------------------------------------------------
// Host-side launch
// ---------------------------------------------------------------------------
extern "C" void kernel_launch(void* const* d_in, const int* in_sizes, int n_in,
                              void* d_out, int out_size, void* d_ws, size_t ws_size,
                              hipStream_t stream) {
    (void)in_sizes; (void)n_in; (void)out_size; (void)ws_size;

    const float* feat = (const float*)d_in[0];
    const float* bw0 = (const float*)d_in[1];  const float* bb0 = (const float*)d_in[2];
    const float* bw1 = (const float*)d_in[3];  const float* bb1 = (const float*)d_in[4];
    const float* bw2 = (const float*)d_in[5];  const float* bb2 = (const float*)d_in[6];
    const float* cw0 = (const float*)d_in[7];  const float* cb0 = (const float*)d_in[8];
    const float* cw1 = (const float*)d_in[9];  const float* cb1 = (const float*)d_in[10];
    const float* cw2 = (const float*)d_in[11]; const float* cb2 = (const float*)d_in[12];

    // Output layout (return order, flat): bbox_x, cls_x, ori, obj_idxes
    float* out_bbox = (float*)d_out;                       // MTOT*4
    float* out_cls  = out_bbox + (size_t)MTOT * 4;         // MTOT*91
    float* out_ori  = out_cls + (size_t)MTOT * CC;         // MTOT*4
    int*   out_idx  = (int*)(out_ori + (size_t)MTOT * 4);  // MTOT

    // Workspace layout
    char* wsb = (char*)d_ws;
    const size_t SZ_BIG = (size_t)512 * 512 * 2 * 2;       // 1 MB per big weight
    unsigned int* sw_b0 = (unsigned int*)(wsb + 0 * SZ_BIG);
    unsigned int* sw_b1 = (unsigned int*)(wsb + 1 * SZ_BIG);
    unsigned int* sw_c0 = (unsigned int*)(wsb + 2 * SZ_BIG);
    unsigned int* sw_c1 = (unsigned int*)(wsb + 3 * SZ_BIG);
    unsigned int* sw_b2 = (unsigned int*)(wsb + 4 * SZ_BIG);                 // 512x16
    unsigned int* sw_c2 = (unsigned int*)(wsb + 4 * SZ_BIG + 32768);         // 512x96
    size_t off = 4 * SZ_BIG + 32768 + (size_t)512 * 96 * 2 * 2;
    off = (off + 255) & ~(size_t)255;
    const size_t SZ_ASPLIT = (size_t)CHUNK * 512 * 2 * 2;  // 16 MB (hi+lo bf16)
    unsigned int* A0 = (unsigned int*)(wsb + off);                  // split(X)
    unsigned int* A1 = (unsigned int*)(wsb + off + SZ_ASPLIT);      // split(hidden)
    float*        H  = (float*)(wsb + off + 2 * SZ_ASPLIT);         // CHUNK x 512 fp32
    float*   scores  = (float*)(wsb + off + 2 * SZ_ASPLIT + (size_t)CHUNK * 512 * 4);

    // 1) Swizzle weights to WMMA B-fragment layout (bf16 hi/lo)
    swizzle_weight_kernel<<<64, 256, 0, stream>>>(bw0, sw_b0, 512, 512);
    swizzle_weight_kernel<<<64, 256, 0, stream>>>(bw1, sw_b1, 512, 512);
    swizzle_weight_kernel<<<64, 256, 0, stream>>>(cw0, sw_c0, 512, 512);
    swizzle_weight_kernel<<<64, 256, 0, stream>>>(cw1, sw_c1, 512, 512);
    swizzle_weight_kernel<<<2,  256, 0, stream>>>(bw2, sw_b2, 4, 16);
    swizzle_weight_kernel<<<12, 256, 0, stream>>>(cw2, sw_c2, CC, 96);

    // 2) Chunked fused MLPs
    const dim3 blk256(256), blk128(128);
    const dim3 gHid(CHUNK / 64, 4);     // NT=32 tiles, 8 per block
    const dim3 gFin(CHUNK / 64, 1);
    const int  gSplit = (CHUNK / 16) * KT_N * 32 / 256;    // 1024 blocks
    for (int c = 0; c < NCHUNK; ++c) {
        const float* Xc = feat + (size_t)c * CHUNK * DD;
        split_acts_kernel<<<gSplit, blk256, 0, stream>>>(Xc, A0);  // shared by both heads

        // bbox head
        gemm_bf16x3_t<4, 2, 0><<<gHid, blk256, 0, stream>>>(A0, sw_b0, bb0, H, 512, 32, 512);
        split_acts_kernel<<<gSplit, blk256, 0, stream>>>(H, A1);
        gemm_bf16x3_t<4, 2, 0><<<gHid, blk256, 0, stream>>>(A1, sw_b1, bb1, H, 512, 32, 512);
        split_acts_kernel<<<gSplit, blk256, 0, stream>>>(H, A1);
        gemm_bf16x3_t<1, 1, 1><<<gFin, blk128, 0, stream>>>(A1, sw_b2, bb2,
                                 out_bbox + (size_t)c * CHUNK * 4, 4, 1, 4);
        // cls head
        gemm_bf16x3_t<4, 2, 0><<<gHid, blk256, 0, stream>>>(A0, sw_c0, cb0, H, 512, 32, 512);
        split_acts_kernel<<<gSplit, blk256, 0, stream>>>(H, A1);
        gemm_bf16x3_t<4, 2, 0><<<gHid, blk256, 0, stream>>>(A1, sw_c1, cb1, H, 512, 32, 512);
        split_acts_kernel<<<gSplit, blk256, 0, stream>>>(H, A1);
        gemm_bf16x3_t<3, 2, 1><<<gFin, blk256, 0, stream>>>(A1, sw_c2, cb2,
                                 out_cls + (size_t)c * CHUNK * CC, CC, 6, CC);
    }

    // 3) Epilogues
    ori_kernel<<<MTOT / 256, 256, 0, stream>>>(out_bbox, out_ori);
    scores_kernel<<<MTOT / 256, 256, 0, stream>>>(out_cls, scores);
    track_kernel<<<BB, QQ, 0, stream>>>(scores, out_idx);
}
